// SheafOptimizedAttention_83270825935580
// MI455X (gfx1250) — compile-verified, matrix operand order checked by reading, not run on hardware
//
#include <hip/hip_runtime.h>

typedef __bf16 bf16_t;
typedef __attribute__((ext_vector_type(16))) __bf16 v16bf;
typedef __attribute__((ext_vector_type(8)))  float  v8f;

#define S 256
#define E 128
#define H 8
#define D 16

#if defined(__gfx1250__) && __has_builtin(__builtin_amdgcn_global_load_async_to_lds_b32) && \
    __has_builtin(__builtin_amdgcn_s_wait_asynccnt)
#define HAVE_ASYNC_LDS 1
typedef __attribute__((address_space(1))) int* gptr_i32;
typedef __attribute__((address_space(3))) int* lptr_i32;
#else
#define HAVE_ASYNC_LDS 0
#endif

// ---------------------------------------------------------------------------
// Elementwise f32 -> bf16 pack
// ---------------------------------------------------------------------------
__global__ void pack_bf16_kernel(const float* __restrict__ src,
                                 bf16_t* __restrict__ dst, int n) {
  int i = blockIdx.x * blockDim.x + threadIdx.x;
  if (i < n) dst[i] = (bf16_t)src[i];
}

// bq|bk|bv -> contiguous [3][128] f32
__global__ void pack_bias3_kernel(const float* __restrict__ a,
                                  const float* __restrict__ b,
                                  const float* __restrict__ c,
                                  float* __restrict__ dst) {
  int i = threadIdx.x; // 128 threads
  dst[i] = a[i]; dst[128 + i] = b[i]; dst[256 + i] = c[i];
}

// ---------------------------------------------------------------------------
// WMMA fragment loaders, bf16 16x16x32 (ISA 7.12.2 layouts)
// A 16x32 (MxK): lane 0-15 -> M, lane/16 selects K-half-of-8;
//   VGPR v, pair p: k = 16*(v/4) + 8*(lane/16) + 2*(v%4) + p
// Per lane this is two contiguous 16-byte runs -> compiler emits b128 loads.
// ---------------------------------------------------------------------------
__device__ inline v16bf load_frag_a16(const bf16_t* __restrict__ X, int ld,
                                      int m0, int k0, int lane) {
  const int m = m0 + (lane & 15);
  const int half = lane >> 4;
  const bf16_t* row = X + (size_t)m * ld;
  v16bf a;
#pragma unroll
  for (int i = 0; i < 16; ++i) {
    const int v = i >> 1, p = i & 1;
    const int k = k0 + ((v >> 2) << 4) + (half << 3) + ((v & 3) << 1) + p;
    a[i] = row[k];
  }
  return a;
}

// B 32x16 (KxN), fed from row-major W[N][K] (so B[k][n] = W[n][k]).
// Assumed to mirror the A layout with N striped across lanes.
__device__ inline v16bf load_frag_b16(const bf16_t* __restrict__ W, int ld,
                                      int n0, int k0, int lane) {
  const int n = n0 + (lane & 15);
  const int half = lane >> 4;
  const bf16_t* row = W + (size_t)n * ld;
  v16bf b;
#pragma unroll
  for (int i = 0; i < 16; ++i) {
    const int v = i >> 1, p = i & 1;
    const int k = k0 + ((v >> 2) << 4) + (half << 3) + ((v & 3) << 1) + p;
    b[i] = row[k];
  }
  return b;
}

// ---------------------------------------------------------------------------
// Generic batched GEMM:  Y[z] = X[z] (MxK) * W[z] (NxK)^T + bias[z]   (f32 out)
// One wave computes a 16 x (16*NT) strip: the A fragment is loaded once per
// K-step and reused across NT accumulators. NT=1 for tiny-N / low-parallelism
// launches, NT=4 where there are plenty of wave-tasks. K % 32 == 0.
// ---------------------------------------------------------------------------
template <int NT>
__global__ void wmma_gemm_xwt_kernel(const bf16_t* __restrict__ Xb,
                                     const bf16_t* __restrict__ Wb,
                                     const float* __restrict__ bias,
                                     float* __restrict__ Yb,
                                     int M, int N, int K,
                                     int ldX, int ldW, int ldY,
                                     long sX, long sW, long sY, long sB) {
  const int wave = threadIdx.x >> 5;
  const int lane = threadIdx.x & 31;
  const int wavesPerBlock = blockDim.x >> 5;
  const int nGroups = N / (16 * NT);
  const int tilesM = M >> 4;
  const int task = blockIdx.x * wavesPerBlock + wave;
  if (task >= tilesM * nGroups) return;           // wave-uniform: EXEC stays all-1s
  const int tm = task / nGroups, tg = task % nGroups;
  const long z = blockIdx.z;
  const bf16_t* X = Xb + z * sX;
  const bf16_t* W = Wb + z * sW;
  float* Y = Yb + z * sY;

  v8f c[NT] = {};
  for (int k0 = 0; k0 < K; k0 += 32) {
    if (k0 + 32 < K)  // locality 3 -> WGP-scope prefetch of next A K-slice
      __builtin_prefetch(&X[(size_t)(tm * 16 + (lane & 15)) * ldX + k0 + 32], 0, 3);
    v16bf a = load_frag_a16(X, ldX, tm * 16, k0, lane);
#pragma unroll
    for (int t = 0; t < NT; ++t) {
      v16bf b = load_frag_b16(W, ldW, (tg * NT + t) * 16, k0, lane);
      c[t] = __builtin_amdgcn_wmma_f32_16x16x32_bf16(
          /*neg_a=*/false, a, /*neg_b=*/false, b,
          /*c_mod=*/(short)0, c[t], /*reuse_a=*/false, /*reuse_b=*/false);
    }
  }

  const int mBase = tm * 16 + ((lane >> 4) << 3);  // C/D: VGPR r -> M = r + 8*(lane/16)
#pragma unroll
  for (int t = 0; t < NT; ++t) {
    const int n = (tg * NT + t) * 16 + (lane & 15);
    const float bval = bias ? bias[z * sB + n] : 0.0f;
#pragma unroll
    for (int r = 0; r < 8; ++r)
      Y[(size_t)(mBase + r) * ldY + n] = c[t][r] + bval;
  }
}

// ---------------------------------------------------------------------------
// Stage f32 [S][E] head-slices into bf16 [H][S][32], zero-padded K (16->32)
// ---------------------------------------------------------------------------
__global__ void stage_head_pad_kernel(const float* __restrict__ src,
                                      bf16_t* __restrict__ dst) {
  int i = blockIdx.x * blockDim.x + threadIdx.x;   // H*S*32 = 65536
  if (i >= H * S * 32) return;
  const int j = i & 31, s = (i >> 5) & (S - 1), h = i >> 13;
  const float v = (j < D) ? src[s * E + h * D + j] : 0.0f;
  dst[i] = (bf16_t)v;
}

// V f32 [S][E] -> bf16 Vt [H][D][S] (transposed so attn@V fits X*W^T form)
__global__ void stage_vt_kernel(const float* __restrict__ V,
                                bf16_t* __restrict__ dst) {
  int i = blockIdx.x * blockDim.x + threadIdx.x;   // H*D*S = 32768
  if (i >= H * D * S) return;
  const int s = i & 255, d = (i >> 8) & 15, h = i >> 12;
  dst[i] = (bf16_t)V[s * E + h * D + d];
}

// ---------------------------------------------------------------------------
// A[h,s,j] = Wt1[j,0:16] . Q[h,s,:]     Bk[h,s,j] = Wt1[j,16:32] . K[h,s,:] + bt1[j]
// ---------------------------------------------------------------------------
__global__ void ab_kernel(const float* __restrict__ Q, const float* __restrict__ Kp,
                          const float* __restrict__ Wt1, const float* __restrict__ bt1,
                          float* __restrict__ A, float* __restrict__ Bk) {
  int i = blockIdx.x * blockDim.x + threadIdx.x;   // H*S*32 = 65536
  if (i >= H * S * 32) return;
  const int j = i & 31, s = (i >> 5) & 255, h = i >> 13;
  const float* wr = Wt1 + j * 32;
  const float* q = Q + s * E + h * D;
  const float* k = Kp + s * E + h * D;
  float a = 0.0f, b = bt1[j];
#pragma unroll
  for (int d = 0; d < D; ++d) { a += wr[d] * q[d]; b += wr[D + d] * k[d]; }
  A[i] = a; Bk[i] = b;
}

// ---------------------------------------------------------------------------
// Hbar[h,k,j] = mean_q tanh(A[h,q,j] + Bk[h,k,j])  -> bf16 [H*S][32]
// Dominant kernel: 16.7M tanh (TRANS pipe). 8 waves/block, LDS tree reduce.
// ---------------------------------------------------------------------------
__global__ void hbar_kernel(const float* __restrict__ A,
                            const float* __restrict__ Bk,
                            bf16_t* __restrict__ Hbar) {
  const int h = blockIdx.x >> 8;                   // blockIdx.x = h*256 + k
  const int tid = threadIdx.x, j = tid & 31, g = tid >> 5;
  const float bkj = Bk[blockIdx.x * 32 + j];
  const float* Ah = A + (h << 13);
  float s = 0.0f;
  for (int q = g; q < S; q += 8) s += tanhf(Ah[q * 32 + j] + bkj);
  __shared__ float red[256];
  red[tid] = s; __syncthreads();
  for (int off = 128; off >= 32; off >>= 1) {
    if (tid < off) red[tid] += red[tid + off];
    __syncthreads();
  }
  if (tid < 32) Hbar[blockIdx.x * 32 + tid] = (bf16_t)(red[tid] * (1.0f / 256.0f));
}

// ---------------------------------------------------------------------------
// K_out[h,k,d] = sum_e G[h*S+k, d*16+e] * K[k, h*16+e]  -> bf16 padded [H][S][32]
// ---------------------------------------------------------------------------
__global__ void kout_kernel(const float* __restrict__ G,
                            const float* __restrict__ Kp,
                            bf16_t* __restrict__ KoutBf) {
  int i = blockIdx.x * blockDim.x + threadIdx.x;   // H*S*32 = 65536
  if (i >= H * S * 32) return;
  const int d = i & 31, k = (i >> 5) & 255, h = i >> 13;
  float val = 0.0f;
  if (d < D) {
    const float* g = G + (size_t)(h * S + k) * 256 + d * D;
    const float* kv = Kp + k * E + h * D;
#pragma unroll
    for (int e = 0; e < D; ++e) val += g[e] * kv[e];
  }
  KoutBf[i] = (bf16_t)val;
}

// ---------------------------------------------------------------------------
// Row softmax over 256 with 1/sqrt(D)=0.25 scale; f32 in, bf16 out.
// Score row is pulled into LDS via the CDNA5 async Global->LDS path.
// ---------------------------------------------------------------------------
__global__ void softmax_kernel(const float* __restrict__ scores,
                               bf16_t* __restrict__ attn) {
  const int row = blockIdx.x;                      // H*S rows
  const int tid = threadIdx.x;                     // 256
  __shared__ float red[256];
#if HAVE_ASYNC_LDS
  __shared__ float rowbuf[256];
  {
    __builtin_amdgcn_global_load_async_to_lds_b32(
        (gptr_i32)(scores + (size_t)row * S + tid),
        (lptr_i32)&rowbuf[tid], 0, 0);
    __builtin_amdgcn_s_wait_asynccnt(0);
  }
  __syncthreads();
  const float v = rowbuf[tid] * 0.25f;
#else
  const float v = scores[(size_t)row * S + tid] * 0.25f;
#endif
  red[tid] = v; __syncthreads();
  for (int off = 128; off > 0; off >>= 1) {
    if (tid < off) red[tid] = fmaxf(red[tid], red[tid + off]);
    __syncthreads();
  }
  const float m = red[0]; __syncthreads();
  const float e = __expf(v - m);
  red[tid] = e; __syncthreads();
  for (int off = 128; off > 0; off >>= 1) {
    if (tid < off) red[tid] += red[tid + off];
    __syncthreads();
  }
  attn[(size_t)row * S + tid] = (bf16_t)(e / red[0]);
}

// ---------------------------------------------------------------------------
extern "C" void kernel_launch(void* const* d_in, const int* in_sizes, int n_in,
                              void* d_out, int out_size, void* d_ws, size_t ws_size,
                              hipStream_t stream) {
  const float* x   = (const float*)d_in[0];
  const float* Wq  = (const float*)d_in[1];
  const float* bq  = (const float*)d_in[2];
  const float* Wk  = (const float*)d_in[3];
  const float* bk  = (const float*)d_in[4];
  const float* Wv  = (const float*)d_in[5];
  const float* bv  = (const float*)d_in[6];
  const float* Wo  = (const float*)d_in[7];
  const float* bo  = (const float*)d_in[8];
  const float* Wt1 = (const float*)d_in[9];
  const float* bt1 = (const float*)d_in[10];
  const float* Wt2 = (const float*)d_in[11];
  const float* bt2 = (const float*)d_in[12];
  float* out = (float*)d_out;

  // ---- workspace carving (~7 MB total) ----
  char* p = (char*)d_ws;
  auto carve = [&](size_t bytes) -> char* {
    char* r = p; p += (bytes + 255) & ~(size_t)255; return r;
  };
  bf16_t* xbf     = (bf16_t*)carve(S * E * 2);           // [256][128]
  bf16_t* wbf     = (bf16_t*)carve(4 * E * E * 2);       // Wq|Wk|Wv|Wo
  bf16_t* wt2bf   = (bf16_t*)carve(256 * 32 * 2);
  float*  bqkv    = (float*) carve(3 * E * 4);
  float*  QKV     = (float*) carve(3 * S * E * 4);       // Q|K|V f32 [S][E]
  bf16_t* Qbf     = (bf16_t*)carve(H * S * 32 * 2);      // padded K=32
  bf16_t* KoutBf  = (bf16_t*)carve(H * S * 32 * 2);
  bf16_t* VtBf    = (bf16_t*)carve(H * D * S * 2);
  float*  Abuf    = (float*) carve(H * S * 32 * 4);
  float*  Bkbuf   = (float*) carve(H * S * 32 * 4);
  bf16_t* HbarBf  = (bf16_t*)carve(H * S * 32 * 2);
  float*  G       = (float*) carve((size_t)H * S * 256 * 4);   // 2 MB
  float*  scores  = (float*) carve((size_t)H * S * S * 4);     // 2 MB
  bf16_t* attnBf  = (bf16_t*)carve((size_t)H * S * S * 2);     // 1 MB
  float*  outHead = (float*) carve(S * E * 4);
  bf16_t* outHbf  = (bf16_t*)carve(S * E * 2);

  const int TPB = 256;
  auto blocks = [](int n, int t) { return (n + t - 1) / t; };

  // ---- 1) pack inputs to bf16 ----
  pack_bf16_kernel<<<blocks(S * E, TPB), TPB, 0, stream>>>(x, xbf, S * E);
  pack_bf16_kernel<<<blocks(E * E, TPB), TPB, 0, stream>>>(Wq, wbf + 0 * E * E, E * E);
  pack_bf16_kernel<<<blocks(E * E, TPB), TPB, 0, stream>>>(Wk, wbf + 1 * E * E, E * E);
  pack_bf16_kernel<<<blocks(E * E, TPB), TPB, 0, stream>>>(Wv, wbf + 2 * E * E, E * E);
  pack_bf16_kernel<<<blocks(E * E, TPB), TPB, 0, stream>>>(Wo, wbf + 3 * E * E, E * E);
  pack_bf16_kernel<<<blocks(256 * 32, TPB), TPB, 0, stream>>>(Wt2, wt2bf, 256 * 32);
  pack_bias3_kernel<<<1, 128, 0, stream>>>(bq, bk, bv, bqkv);

  // ---- 2) fused QKV projection: z=3 over weights, X shared (NT=1: keep waves) --
  {
    int tasks = (S / 16) * (E / 16);               // 128 per z
    wmma_gemm_xwt_kernel<1><<<dim3(blocks(tasks, 4), 1, 3), 128, 0, stream>>>(
        xbf, wbf, bqkv, QKV, S, E, E, E, E, E,
        /*sX=*/0, /*sW=*/(long)E * E, /*sY=*/(long)S * E, /*sB=*/E);
  }
  const float* Qf = QKV;
  const float* Kf = QKV + S * E;
  const float* Vf = QKV + 2 * S * E;

  // ---- 3) staging + transport-MLP precomputation ----
  stage_head_pad_kernel<<<blocks(H * S * 32, TPB), TPB, 0, stream>>>(Qf, Qbf);
  stage_vt_kernel<<<blocks(H * D * S, TPB), TPB, 0, stream>>>(Vf, VtBf);
  ab_kernel<<<blocks(H * S * 32, TPB), TPB, 0, stream>>>(Qf, Kf, Wt1, bt1, Abuf, Bkbuf);

  // ---- 4) Hbar = mean_q tanh(A[q]+B[k])  (dominant TRANS work) ----
  hbar_kernel<<<H * S, 256, 0, stream>>>(Abuf, Bkbuf, HbarBf);

  // ---- 5) G = Hbar (2048x32) * Wt2^T (32x256) + bt2  (NT=4: A reuse, 512 waves) --
  {
    int tasks = (H * S / 16) * (256 / 64);         // 512
    wmma_gemm_xwt_kernel<4><<<dim3(blocks(tasks, 4), 1, 1), 128, 0, stream>>>(
        HbarBf, wt2bf, bt2, G, H * S, 256, 32, 32, 32, 256, 0, 0, 0, 0);
  }

  // ---- 6) K_out[h,k] = reshape(G) . K[h,k]  -> padded bf16 ----
  kout_kernel<<<blocks(H * S * 32, TPB), TPB, 0, stream>>>(G, Kf, KoutBf);

  // ---- 7) scores = Q . K_out^T per head (K padded to 32; NT=4, 8x64 waves) ----
  {
    int tasks = (S / 16) * (S / 64);               // 64 per z
    wmma_gemm_xwt_kernel<4><<<dim3(blocks(tasks, 4), 1, H), 128, 0, stream>>>(
        Qbf, KoutBf, nullptr, scores, S, S, 32, 32, 32, S,
        /*sX=*/(long)S * 32, /*sW=*/(long)S * 32, /*sY=*/(long)S * S, 0);
  }

  // ---- 8) softmax (scale 1/4) -> bf16 attn ----
  softmax_kernel<<<H * S, S, 0, stream>>>(scores, attnBf);

  // ---- 9) out_head = attn (256x256) . V (256x16), written into [S][E] slices --
  {
    int tasks = (S / 16) * (D / 16);               // 16 per z (NT=1: N is one tile)
    wmma_gemm_xwt_kernel<1><<<dim3(blocks(tasks, 4), 1, H), 128, 0, stream>>>(
        attnBf, VtBf, nullptr, outHead, S, D, S, S, S, E,
        /*sX=*/(long)S * S, /*sW=*/(long)D * S, /*sY=*/D, 0);
  }

  // ---- 10) final projection: out = out_head . Wo^T + bo ----
  pack_bf16_kernel<<<blocks(S * E, TPB), TPB, 0, stream>>>(outHead, outHbf, S * E);
  {
    int tasks = (S / 16) * (E / 16);               // 128
    wmma_gemm_xwt_kernel<1><<<dim3(blocks(tasks, 4), 1, 1), 128, 0, stream>>>(
        outHbf, wbf + 3 * E * E, bo, out, S, E, E, E, E, E, 0, 0, 0, 0);
  }
}
